// Me_1219770712245
// MI455X (gfx1250) — matrix operand order repaired and checked by reading, not compiled, not run on hardware
//
#include <hip/hip_runtime.h>
#include <hip/hip_bf16.h>

typedef __attribute__((ext_vector_type(16))) _Float16 v16h;
typedef __attribute__((ext_vector_type(8)))  _Float16 v8h;
typedef __attribute__((ext_vector_type(8)))  float    v8f;

#define BATCH 32

// ---------------------------------------------------------------------------
// Prep kernels
// ---------------------------------------------------------------------------
__global__ void cvt_f16_kernel(const float* __restrict__ src, _Float16* __restrict__ dst, int n) {
    int i = blockIdx.x * blockDim.x + threadIdx.x;
    if (i < n) dst[i] = (_Float16)src[i];
}

// bas_w [64][64][3][3] -> tap-major f16 [9][64][64]
__global__ void bas_reorg_kernel(const float* __restrict__ w, _Float16* __restrict__ wh) {
    int i = blockIdx.x * blockDim.x + threadIdx.x;
    if (i >= 9 * 64 * 64) return;
    int t = i / 4096, rem = i % 4096;
    int c2 = rem / 64, c1 = rem % 64;
    wh[i] = (_Float16)w[(c2 * 64 + c1) * 9 + t];
}

// p = [gamma,beta,mean,var] each [C]; sb = [scale[C], bias[C]]
__global__ void bn_prep_kernel(const float* __restrict__ p, float* __restrict__ sb, int C) {
    int c = blockIdx.x * blockDim.x + threadIdx.x;
    if (c >= C) return;
    float g = p[c], b = p[C + c], m = p[2 * C + c], v = p[3 * C + c];
    float s = g * rsqrtf(v + 1e-5f);
    sb[c] = s;
    sb[C + c] = b - m * s;
}

// ---------------------------------------------------------------------------
// Spatial attention
// ---------------------------------------------------------------------------
__global__ void chmax_kernel(const float* __restrict__ in, float* __restrict__ m, int C, int HW) {
    int b = blockIdx.y;
    int p = blockIdx.x * blockDim.x + threadIdx.x;
    if (p >= HW) return;
    const float* ip = in + (size_t)b * C * HW + p;
    float mx = ip[0];
    for (int c = 1; c < C; ++c) mx = fmaxf(mx, ip[(size_t)c * HW]);
    m[(size_t)b * HW + p] = mx;
}

__global__ void saconv_kernel(const float* __restrict__ m, const float* __restrict__ w,
                              float* __restrict__ s, int H, int Ks, int pad) {
    int b = blockIdx.y;
    int HW = H * H;
    int p = blockIdx.x * blockDim.x + threadIdx.x;
    if (p >= HW) return;
    int y = p / H, x = p % H;
    const float* mp = m + (size_t)b * HW;
    float acc = 0.f;
    for (int i = 0; i < Ks; ++i) {
        int yy = y + i - pad;
        if ((unsigned)yy >= (unsigned)H) continue;
        for (int j = 0; j < Ks; ++j) {
            int xx = x + j - pad;
            if ((unsigned)xx < (unsigned)H) acc += w[i * Ks + j] * mp[yy * H + xx];
        }
    }
    s[(size_t)b * HW + p] = 1.f / (1.f + __expf(-acc));
}

__global__ void gate_kernel(const float* __restrict__ r, const float* __restrict__ s,
                            float* __restrict__ rd, int CHW, int HW, size_t total) {
    size_t i = (size_t)blockIdx.x * blockDim.x + threadIdx.x;
    if (i >= total) return;
    size_t b = i / (size_t)CHW;
    size_t p = i % (size_t)HW;
    rd[i] = r[i] * (1.f + s[b * HW + p]);
}

// ---------------------------------------------------------------------------
// WMMA GEMM for 1x1 conv + fused BN:  out[b][M][N] = BN( Wh[M][K] @ X[b][K][N] )
// 128-thread block = 4 waves. One shared 32x16 B tile per k-step, staged
// cooperatively (each thread loads one float4); wave w computes M-tile
// blockIdx.y*64 + w*16, so the staged tile is reused by 4 WMMA tiles.
// Requires K % 32 == 0, N % 16 == 0, M % 64 == 0 per grid.y slice.
// ---------------------------------------------------------------------------
__global__ __launch_bounds__(128)
void gemm1x1_bn_kernel(const float* __restrict__ X, const _Float16* __restrict__ Wh,
                       const float* __restrict__ bnsb, float* __restrict__ out,
                       int M, int N, int K)
{
    __shared__ _Float16 lds[32 * 16];          // transposed tile: lds[n*32 + k]
    const int tid  = threadIdx.x;
    const int wave = tid >> 5;
    const int lane = tid & 31;
    const int hi   = (lane >> 4);              // K-half selector within fragment
    const int l16  = lane & 15;
    const int n0 = blockIdx.x * 16;
    const int m0 = blockIdx.y * 64 + wave * 16;
    const int b  = blockIdx.z;
    const float* Xb = X + (size_t)b * K * N;

    const int row = tid >> 2;                  // 0..31: K-row within tile
    const int q   = tid & 3;                   // quarter of the 16-wide N strip

    v8f acc = {};
    for (int kt = 0; kt < K; kt += 32) {
        // Stage B tile [K=32][N=16] -> LDS transposed [N][K] (f32 -> f16)
        float4 v = *(const float4*)(Xb + (size_t)(kt + row) * N + n0 + q * 4);
        lds[(q * 4 + 0) * 32 + row] = (_Float16)v.x;
        lds[(q * 4 + 1) * 32 + row] = (_Float16)v.y;
        lds[(q * 4 + 2) * 32 + row] = (_Float16)v.z;
        lds[(q * 4 + 3) * 32 + row] = (_Float16)v.w;
        __syncthreads();

        // A fragment (16-bit A 16x32 layout): M = m0+l16; K = hi*8+{0..7} and 16+hi*8+{0..7}
        const _Float16* wrow = Wh + (size_t)(m0 + l16) * K + kt;
        if (kt + 32 < K) __builtin_prefetch(wrow + 32, 0, 1);
        v8h alo = *(const v8h*)(wrow + hi * 8);
        v8h ahi = *(const v8h*)(wrow + 16 + hi * 8);
        v16h a;
        #pragma unroll
        for (int i = 0; i < 8; ++i) { a[i] = alo[i]; a[8 + i] = ahi[i]; }

        // B fragment: N = l16, K = hi*8+{0..7} and 16+hi*8+{0..7}
        const v8h* brow = (const v8h*)(&lds[l16 * 32 + hi * 8]);
        v8h blo = brow[0], bhi2 = brow[2];
        v16h bf;
        #pragma unroll
        for (int i = 0; i < 8; ++i) { bf[i] = blo[i]; bf[8 + i] = bhi2[i]; }

        acc = __builtin_amdgcn_wmma_f32_16x16x32_f16(false, a, false, bf,
                                                     (short)0, acc, false, false);
        __syncthreads();
    }

    // Epilogue: BN + store.  C/D layout: element r -> M = hi*8 + r, N = l16.
    float* ob = out + (size_t)b * M * N;
    #pragma unroll
    for (int r = 0; r < 8; ++r) {
        int m = m0 + hi * 8 + r;
        ob[(size_t)m * N + n0 + l16] = acc[r] * bnsb[m] + bnsb[M + m];
    }
}

// ---------------------------------------------------------------------------
// WMMA implicit-GEMM 3x3 conv on rd0 [B][64][64][64] with fused epilogue:
//   res = BN(conv) * x3 + rd0
// 128-thread block = 4 waves covering all 64 output channels; one shared
// (zero-padded, shifted) B tile per (tap, k-half) step.
// ---------------------------------------------------------------------------
__global__ __launch_bounds__(128)
void conv3x3_wmma_kernel(const float* __restrict__ Xin, const _Float16* __restrict__ Wh9,
                         const float* __restrict__ bnsb, const float* __restrict__ x3,
                         float* __restrict__ res)
{
    __shared__ _Float16 lds[32 * 16];
    const int tid  = threadIdx.x;
    const int wave = tid >> 5;
    const int lane = tid & 31;
    const int hi   = (lane >> 4);
    const int l16  = lane & 15;
    const int n0 = blockIdx.x * 16;            // pixel tile (0..4095)
    const int m0 = wave * 16;                  // out-channel tile
    const int b  = blockIdx.z;
    const int y  = n0 >> 6;
    const int x0 = n0 & 63;
    const float* Xb = Xin + (size_t)b * 64 * 64 * 64;

    const int row = tid >> 2;                  // 0..31: input channel within k-half
    const int q   = tid & 3;

    v8f acc = {};
    for (int t = 0; t < 9; ++t) {
        int dy = t / 3 - 1, dx = t % 3 - 1;
        int yy = y + dy;
        bool yok = (unsigned)yy < 64u;
        for (int kh = 0; kh < 2; ++kh) {
            int c1 = kh * 32 + row;
            const float* src = Xb + ((size_t)c1 * 64 + (yok ? yy : 0)) * 64;
            #pragma unroll
            for (int j = 0; j < 4; ++j) {
                int n = q * 4 + j;
                int xx = x0 + n + dx;
                float v = (yok && (unsigned)xx < 64u) ? src[xx] : 0.f;
                lds[n * 32 + row] = (_Float16)v;
            }
            __syncthreads();

            const _Float16* wrow = Wh9 + (size_t)t * 4096 + (m0 + l16) * 64 + kh * 32;
            v8h alo = *(const v8h*)(wrow + hi * 8);
            v8h ahi = *(const v8h*)(wrow + 16 + hi * 8);
            v16h a;
            #pragma unroll
            for (int i = 0; i < 8; ++i) { a[i] = alo[i]; a[8 + i] = ahi[i]; }

            const v8h* brow = (const v8h*)(&lds[l16 * 32 + hi * 8]);
            v8h blo = brow[0], bhi2 = brow[2];
            v16h bf;
            #pragma unroll
            for (int i = 0; i < 8; ++i) { bf[i] = blo[i]; bf[8 + i] = bhi2[i]; }

            acc = __builtin_amdgcn_wmma_f32_16x16x32_f16(false, a, false, bf,
                                                         (short)0, acc, false, false);
            __syncthreads();
        }
    }

    size_t base = (size_t)b * 64 * 4096;
    #pragma unroll
    for (int r = 0; r < 8; ++r) {
        int m = m0 + hi * 8 + r;
        size_t idx = base + (size_t)m * 4096 + n0 + l16;
        float v = acc[r] * bnsb[m] + bnsb[64 + m];
        res[idx] = v * x3[idx] + Xin[idx];
    }
}

// ---------------------------------------------------------------------------
// 2x2 average pool (H even). total = B*C*(H/2)*(H/2)
// ---------------------------------------------------------------------------
__global__ void pool2_kernel(const float* __restrict__ in, float* __restrict__ out,
                             int H, int total) {
    int i = blockIdx.x * blockDim.x + threadIdx.x;
    if (i >= total) return;
    int Ho = H / 2;
    int xo = i % Ho; int t = i / Ho;
    int yo = t % Ho; int bc = t / Ho;
    const float* ip = in + ((size_t)bc * H + yo * 2) * H + xo * 2;
    out[i] = 0.25f * (ip[0] + ip[1] + ip[H] + ip[H + 1]);
}

// ---------------------------------------------------------------------------
// Per-sample dynamic depthwise dilated conv (dil=2); sample 0 passes through.
// One block per (b,c). The S*S kernel weights are copied global->LDS with the
// CDNA5 async-DMA path (global_load_async_to_lds_b128 + s_wait_asynccnt).
// ---------------------------------------------------------------------------
__global__ void dd_apply_kernel(const float* __restrict__ low, const float* __restrict__ ker,
                                float* __restrict__ out, int C, int H, int S, int pad) {
    extern __shared__ float kw[];
    int bc = blockIdx.x;
    int b = bc / C;
    int HW = H * H;
    const float* kerp = ker + (size_t)bc * S * S;

    // Async copy of S*S floats (S*S*4 bytes, multiple of 16) into LDS.
    int nvec = (S * S) >> 2;                       // 16-byte chunks
    if ((int)threadIdx.x < nvec) {
        const float* gsrc = kerp + threadIdx.x * 4;
        unsigned ldsaddr = (unsigned)(uintptr_t)kw + threadIdx.x * 16u;
        asm volatile(
            "global_load_async_to_lds_b128 %0, %1, off\n\t"
            "s_wait_asynccnt 0x0"
            :: "v"(ldsaddr), "v"(gsrc) : "memory");
    }
    __syncthreads();

    const float* lp = low + (size_t)bc * HW;
    float* op = out + (size_t)bc * HW;
    for (int p = threadIdx.x; p < HW; p += blockDim.x) {
        if (b == 0) { op[p] = lp[p]; continue; }
        int y = p / H, x = p % H;
        float s = 0.f;
        for (int i = 0; i < S; ++i) {
            int yy = y + 2 * i - pad;
            if ((unsigned)yy >= (unsigned)H) continue;
            const float* row = lp + yy * H;
            for (int j = 0; j < S; ++j) {
                int xx = x + 2 * j - pad;
                if ((unsigned)xx < (unsigned)H) s += kw[i * S + j] * row[xx];
            }
        }
        op[p] = s;
    }
}

// ---------------------------------------------------------------------------
// Final: 1x1 conv 64->1 + BN + nearest x4 upsample, written to both outputs.
// ---------------------------------------------------------------------------
__global__ void end_kernel(const float* __restrict__ res64, const float* __restrict__ w,
                           const float* __restrict__ sb, float* __restrict__ out, int B) {
    int b = blockIdx.y;
    int p = blockIdx.x * blockDim.x + threadIdx.x;
    if (p >= 4096) return;
    int y = p >> 6, x = p & 63;
    const float* rp = res64 + (size_t)b * 64 * 4096 + p;
    float acc = 0.f;
    for (int c = 0; c < 64; ++c) acc += w[c] * rp[(size_t)c * 4096];
    float v = acc * sb[0] + sb[1];
    size_t base = (size_t)b * 65536;
    size_t second = (size_t)B * 65536;
    for (int dy = 0; dy < 4; ++dy) {
        size_t rowo = base + (size_t)(y * 4 + dy) * 256 + x * 4;
        #pragma unroll
        for (int dx = 0; dx < 4; ++dx) {
            out[rowo + dx] = v;
            out[second + rowo + dx] = v;
        }
    }
}

// ---------------------------------------------------------------------------
// Host-side orchestration
// ---------------------------------------------------------------------------
extern "C" void kernel_launch(void* const* d_in, const int* in_sizes, int n_in,
                              void* d_out, int out_size, void* d_ws, size_t ws_size,
                              hipStream_t stream)
{
    (void)in_sizes; (void)n_in; (void)out_size; (void)ws_size;
    const float* r0 = (const float*)d_in[0];
    const float* r1 = (const float*)d_in[1];
    const float* r2 = (const float*)d_in[2];
    const float* r3 = (const float*)d_in[3];
    const float* d0 = (const float*)d_in[4];
    const float* d1 = (const float*)d_in[5];
    const float* d2 = (const float*)d_in[6];
    const float* d3 = (const float*)d_in[7];
    const float* sa1_w = (const float*)d_in[8];
    const float* sa2_w = (const float*)d_in[9];
    const float* sa3_w = (const float*)d_in[10];
    const float* sa4_w = (const float*)d_in[11];
    const float* dd1_w = (const float*)d_in[12];
    const float* dd1_bn = (const float*)d_in[13];
    const float* dd2_w = (const float*)d_in[14];
    const float* dd2_bn = (const float*)d_in[15];
    const float* dd3_w = (const float*)d_in[16];
    const float* dd3_bn = (const float*)d_in[17];
    const float* bas_w = (const float*)d_in[18];
    const float* bas_bn = (const float*)d_in[19];
    const float* end_w = (const float*)d_in[20];
    const float* end_bn = (const float*)d_in[21];

    char* ws = (char*)d_ws;
    size_t off = 0;
    auto alloc = [&](size_t bytes) -> char* {
        char* p = ws + off;
        off = (off + bytes + 255) & ~(size_t)255;
        return p;
    };

    float* rd0  = (float*)alloc((size_t)8388608 * 4);
    float* rd1  = (float*)alloc((size_t)4194304 * 4);
    float* rd2  = (float*)alloc((size_t)2621440 * 4);
    float* rd3  = (float*)alloc((size_t)1048576 * 4);
    float* mbuf = (float*)alloc((size_t)131072 * 4);
    float* sbuf = (float*)alloc((size_t)131072 * 4);
    float* c1b  = (float*)alloc((size_t)655360 * 4);
    float* k1p  = (float*)alloc((size_t)163840 * 4);
    float* x1   = (float*)alloc((size_t)2621440 * 4);
    float* c2b  = (float*)alloc((size_t)1048576 * 4);
    float* k2p  = (float*)alloc((size_t)262144 * 4);
    float* x2   = (float*)alloc((size_t)4194304 * 4);
    float* c3b  = (float*)alloc((size_t)2097152 * 4);
    float* k3p  = (float*)alloc((size_t)524288 * 4);
    float* x3   = (float*)alloc((size_t)8388608 * 4);
    float* res64= (float*)alloc((size_t)8388608 * 4);
    _Float16* dd1h = (_Float16*)alloc((size_t)163840 * 2);
    _Float16* dd2h = (_Float16*)alloc((size_t)40960 * 2);
    _Float16* dd3h = (_Float16*)alloc((size_t)8192 * 2);
    _Float16* bash = (_Float16*)alloc((size_t)36864 * 2);
    float* dd1sb = (float*)alloc(640 * 4);
    float* dd2sb = (float*)alloc(256 * 4);
    float* dd3sb = (float*)alloc(128 * 4);
    float* bassb = (float*)alloc(128 * 4);
    float* endsb = (float*)alloc(2 * 4);

    // --- prep: f16 weights + BN fold ---
    cvt_f16_kernel<<<(163840 + 255) / 256, 256, 0, stream>>>(dd1_w, dd1h, 163840);
    cvt_f16_kernel<<<(40960 + 255) / 256, 256, 0, stream>>>(dd2_w, dd2h, 40960);
    cvt_f16_kernel<<<(8192 + 255) / 256, 256, 0, stream>>>(dd3_w, dd3h, 8192);
    bas_reorg_kernel<<<(36864 + 255) / 256, 256, 0, stream>>>(bas_w, bash);
    bn_prep_kernel<<<2, 256, 0, stream>>>(dd1_bn, dd1sb, 320);
    bn_prep_kernel<<<1, 128, 0, stream>>>(dd2_bn, dd2sb, 128);
    bn_prep_kernel<<<1, 64, 0, stream>>>(dd3_bn, dd3sb, 64);
    bn_prep_kernel<<<1, 64, 0, stream>>>(bas_bn, bassb, 64);
    bn_prep_kernel<<<1, 32, 0, stream>>>(end_bn, endsb, 1);

    // --- spatial attention gating per scale ---
    struct Scale { const float* r; const float* d; const float* w; float* rd; int C, H, Ks, pad; };
    Scale sc[4] = {
        { r0, d0, sa1_w, rd0, 64, 64, 7, 3 },
        { r1, d1, sa2_w, rd1, 128, 32, 7, 3 },
        { r2, d2, sa3_w, rd2, 320, 16, 3, 1 },
        { r3, d3, sa4_w, rd3, 512, 8, 3, 1 },
    };
    for (int i = 0; i < 4; ++i) {
        int HW = sc[i].H * sc[i].H;
        dim3 g((HW + 255) / 256, BATCH);
        chmax_kernel<<<g, 256, 0, stream>>>(sc[i].d, mbuf, sc[i].C, HW);
        saconv_kernel<<<g, 256, 0, stream>>>(mbuf, sc[i].w, sbuf, sc[i].H, sc[i].Ks, sc[i].pad);
        size_t total = (size_t)BATCH * sc[i].C * HW;
        gate_kernel<<<(unsigned)((total + 255) / 256), 256, 0, stream>>>(
            sc[i].r, sbuf, sc[i].rd, sc[i].C * HW, HW, total);
    }

    // --- DD1: 512->320 GEMM on rd3 (N=64), pool 8->4, dyn depthwise on rd2 ---
    gemm1x1_bn_kernel<<<dim3(4, 5, BATCH), 128, 0, stream>>>(rd3, dd1h, dd1sb, c1b, 320, 64, 512);
    pool2_kernel<<<(163840 + 255) / 256, 256, 0, stream>>>(c1b, k1p, 8, 163840);
    dd_apply_kernel<<<BATCH * 320, 256, 4 * 4 * 4, stream>>>(rd2, k1p, x1, 320, 16, 4, 3);

    // --- DD2: 320->128 GEMM on x1 (N=256), pool 16->8, dyn depthwise on rd1 ---
    gemm1x1_bn_kernel<<<dim3(16, 2, BATCH), 128, 0, stream>>>(x1, dd2h, dd2sb, c2b, 128, 256, 320);
    pool2_kernel<<<(262144 + 255) / 256, 256, 0, stream>>>(c2b, k2p, 16, 262144);
    dd_apply_kernel<<<BATCH * 128, 256, 8 * 8 * 4, stream>>>(rd1, k2p, x2, 128, 32, 8, 7);

    // --- DD3: 128->64 GEMM on x2 (N=1024), pool 32->16, dyn depthwise on rd0 ---
    gemm1x1_bn_kernel<<<dim3(64, 1, BATCH), 128, 0, stream>>>(x2, dd3h, dd3sb, c3b, 64, 1024, 128);
    pool2_kernel<<<(524288 + 255) / 256, 256, 0, stream>>>(c3b, k3p, 32, 524288);
    dd_apply_kernel<<<BATCH * 64, 256, 16 * 16 * 4, stream>>>(rd0, k3p, x3, 64, 64, 16, 15);

    // --- bas 3x3 conv (implicit GEMM, WMMA) + fused res = BN(conv)*x3 + rd0 ---
    conv3x3_wmma_kernel<<<dim3(256, 1, BATCH), 128, 0, stream>>>(rd0, bash, bassb, x3, res64);

    // --- end 1x1 + BN + nearest x4 upsample -> both output copies ---
    end_kernel<<<dim3(16, BATCH), 256, 0, stream>>>(res64, end_w, endsb, (float*)d_out, BATCH);
}
